// _SelfAttention_36790689857559
// MI455X (gfx1250) — matrix operand order repaired and checked by reading, not compiled, not run on hardware
//
#include <hip/hip_runtime.h>
#include <hip/hip_bf16.h>
#include <math.h>
#include <stdint.h>

// ---------------------------------------------------------------------------
// CDNA5 (gfx1250) self-attention:
//   - one-time f32->bf16 conversion of x and weights
//   - WMMA bf16 GEMMs with async global->LDS double-buffered staging
//   - flash attention (online softmax) with WMMA for S=QK^T and O=PV,
//     softmax reductions via v_permlane16_b32 butterflies (no LDS traffic)
// wave32; v_wmma_f32_16x16x32_bf16; global_load_async_to_lds_b128 (ASYNCcnt)
// ---------------------------------------------------------------------------

typedef __bf16 bf16;
typedef __attribute__((ext_vector_type(16))) __bf16 v16bf;
typedef __attribute__((ext_vector_type(8)))  float  v8f;

#define D_MODEL  1024
#define N_HEADS  16
#define HEAD_DIM 64
#define SEQ      2048
#define BATCH    4
#define MROWS    (BATCH * SEQ)   // 8192

__device__ __forceinline__ v8f wmma_bf16(v16bf a, v16bf b, v8f c) {
  // 8 args: (neg_a, A, neg_b, B, c_mod, C, reuse_a, reuse_b)
  return __builtin_amdgcn_wmma_f32_16x16x32_bf16(false, a, false, b,
                                                 (short)0, c, false, false);
}

__device__ __forceinline__ void async_copy_b128(unsigned lds_off, const void* gaddr) {
  // GLOBAL_LOAD_ASYNC_TO_LDS_B128: VDST = LDS byte address VGPR, ADDR = 64-bit
  // global address VGPR pair (GV mode). Tracked with ASYNCcnt.
  asm volatile("global_load_async_to_lds_b128 %0, %1, off"
               :: "v"(lds_off), "v"(gaddr) : "memory");
}

__device__ __forceinline__ void wait_async0() {
  asm volatile("s_wait_asynccnt 0x0" ::: "memory");
}

// ---------------------------------------------------------------------------
// Intra-16-lane butterfly combine via v_permlane16_b32 (VALU, no LDS).
// Lanes 0-15 and 16-31 reduce independently (matches C-matrix lane halves).
// Select nibble packs for lane i <- i^mask within each 16-lane row.
// ---------------------------------------------------------------------------
#if __has_builtin(__builtin_amdgcn_permlane16)
__device__ __forceinline__ float xor_lane16(float x, int step) {
  unsigned lo, hi;
  switch (step) {                  // nibble i = source lane for lane i
    case 1: lo = 0x67452301u; hi = 0xEFCDAB89u; break;   // i^1
    case 2: lo = 0x54761032u; hi = 0xDCFE98BAu; break;   // i^2
    case 4: lo = 0x32107654u; hi = 0xBA98FEDCu; break;   // i^4
    default:lo = 0xFEDCBA98u; hi = 0x76543210u; break;   // i^8
  }
  unsigned u = __float_as_uint(x);
  unsigned r = __builtin_amdgcn_permlane16(u, u, lo, hi, false, false);
  return __uint_as_float(r);
}
#else
__device__ __forceinline__ float xor_lane16(float x, int step) {
  return __shfl_xor(x, step, 16);
}
#endif

__device__ __forceinline__ float reduce_max16(float x) {
  x = fmaxf(x, xor_lane16(x, 1));
  x = fmaxf(x, xor_lane16(x, 2));
  x = fmaxf(x, xor_lane16(x, 4));
  x = fmaxf(x, xor_lane16(x, 8));
  return x;
}
__device__ __forceinline__ float reduce_add16(float x) {
  x += xor_lane16(x, 1);
  x += xor_lane16(x, 2);
  x += xor_lane16(x, 4);
  x += xor_lane16(x, 8);
  return x;
}

// ---------------------------------------------------------------------------
// f32 -> bf16 elementwise (x and the four weight matrices; done once)
// ---------------------------------------------------------------------------
__global__ void cvt_bf16(const float* __restrict__ s, bf16* __restrict__ d)
{
  size_t i = ((size_t)blockIdx.x * 256 + threadIdx.x) * 4;
  float4 v = *(const float4*)(s + i);
  d[i + 0] = (bf16)v.x; d[i + 1] = (bf16)v.y;
  d[i + 2] = (bf16)v.z; d[i + 3] = (bf16)v.w;
}

// ---------------------------------------------------------------------------
// GEMM: C[M,E] = Xb[M,D] * Wb[E,D]^T  (einsum 'md,ed->me'), all-bf16 inputs.
//  MODE 0: f32  -> outF[(b*16+h)*SEQ*64 + n*64 + d]   (Q projection)
//  MODE 1: bf16 -> outB same [B,H,N,64] layout        (K / V projections)
//  MODE 2: f32  -> outF[m*D_MODEL + e]                (final output GEMM)
// Block = 128 threads (4 waves). Block tile 128x64 (wave tile 32x64), K step 32.
// ---------------------------------------------------------------------------
template<int MODE>
__global__ __launch_bounds__(128)
void gemm_wmma(const bf16* __restrict__ Xb, const bf16* __restrict__ Wb,
               float* __restrict__ outF, bf16* __restrict__ outB)
{
  // 40-elem rows: 80B (16B-aligned chunks), 20-dword stride -> conflict-free
  __shared__ bf16 lds_a[2][128][40];   // X tile  (rows m, cols k)
  __shared__ bf16 lds_b[2][64][40];    // W tile  (rows e, cols k)

  const int tid  = threadIdx.x;
  const int wv   = tid >> 5;
  const int lane = tid & 31;
  const int half = lane >> 4;
  const int hl   = lane & 15;
  const int m0   = blockIdx.y * 128;
  const int e0   = blockIdx.x * 64;

  auto stage = [&](int k0, int buf) {
    #pragma unroll
    for (int j = 0; j < 4; ++j) {                // A: 512 chunks of 8 bf16
      int q = tid + j * 128;
      int r = q >> 2, dc = (q & 3) * 8;
      async_copy_b128((unsigned)(uintptr_t)&lds_a[buf][r][dc],
                      Xb + (size_t)(m0 + r) * D_MODEL + k0 + dc);
    }
    #pragma unroll
    for (int j = 0; j < 2; ++j) {                // B: 256 chunks of 8 bf16
      int q = tid + j * 128;
      int r = q >> 2, dc = (q & 3) * 8;
      async_copy_b128((unsigned)(uintptr_t)&lds_b[buf][r][dc],
                      Wb + (size_t)(e0 + r) * D_MODEL + k0 + dc);
    }
  };

  v8f acc[2][4] = {};                  // wave tile 32x64

  stage(0, 0);
  for (int k0 = 0; k0 < D_MODEL; k0 += 32) {
    const int buf = (k0 >> 5) & 1;
    wait_async0();                     // own async loads landed in LDS
    __syncthreads();                   // all waves' loads visible; prev reads done
    if (k0 + 32 < D_MODEL) stage(k0 + 32, buf ^ 1);

    v16bf a[2];
    #pragma unroll
    for (int s = 0; s < 2; ++s) {
      #pragma unroll
      for (int v = 0; v < 8; ++v) {
        int kk = (v & 3) * 2 + (v >> 2) * 16 + half * 8;
        a[s][2 * v]     = lds_a[buf][wv * 32 + s * 16 + hl][kk];
        a[s][2 * v + 1] = lds_a[buf][wv * 32 + s * 16 + hl][kk + 1];
      }
    }
    #pragma unroll
    for (int t = 0; t < 4; ++t) {
      v16bf b;
      #pragma unroll
      for (int v = 0; v < 8; ++v) {
        int kk = (v & 3) * 2 + (v >> 2) * 16 + half * 8;
        b[2 * v]     = lds_b[buf][t * 16 + hl][kk];
        b[2 * v + 1] = lds_b[buf][t * 16 + hl][kk + 1];
      }
      acc[0][t] = wmma_bf16(a[0], b, acc[0][t]);
      acc[1][t] = wmma_bf16(a[1], b, acc[1][t]);
    }
  }

  // epilogue: C layout -> row = r + 8*half, col = lane%16 (branch-free)
  #pragma unroll
  for (int s = 0; s < 2; ++s) {
    #pragma unroll
    for (int t = 0; t < 4; ++t) {
      #pragma unroll
      for (int r = 0; r < 8; ++r) {
        int m = m0 + wv * 32 + s * 16 + half * 8 + r;
        int e = e0 + t * 16 + hl;
        float val = acc[s][t][r];
        if (MODE == 2) {
          outF[(size_t)m * D_MODEL + e] = val;
        } else {
          int b = m >> 11, n = m & (SEQ - 1);
          int h = e >> 6,  d = e & 63;
          size_t off = ((size_t)(b * N_HEADS + h) * SEQ + n) * HEAD_DIM + d;
          if (MODE == 0) outF[off] = val;
          else           outB[off] = (bf16)val;
        }
      }
    }
  }
}

// ---------------------------------------------------------------------------
// RoPE: pair (j, j+32), angle = n * 10000^(-j/32)
// ---------------------------------------------------------------------------
__global__ void rotary_f32(float* __restrict__ Q)
{
  unsigned idx = blockIdx.x * 256u + threadIdx.x;   // BATCH*H*SEQ*32 threads
  int j = idx & 31;
  unsigned rest = idx >> 5;
  int n = rest & (SEQ - 1);
  unsigned bh = rest >> 11;
  float* p = Q + ((size_t)bh * SEQ + n) * HEAD_DIM;
  float fr = __powf(10000.f, -(float)j * (1.f / 32.f));
  float s, c;
  __sincosf((float)n * fr, &s, &c);
  float x1 = p[j], x2 = p[j + 32];
  p[j]      = x1 * c - x2 * s;
  p[j + 32] = x1 * s + x2 * c;
}

__global__ void rotary_bf16(bf16* __restrict__ K)
{
  unsigned idx = blockIdx.x * 256u + threadIdx.x;
  int j = idx & 31;
  unsigned rest = idx >> 5;
  int n = rest & (SEQ - 1);
  unsigned bh = rest >> 11;
  bf16* p = K + ((size_t)bh * SEQ + n) * HEAD_DIM;
  float fr = __powf(10000.f, -(float)j * (1.f / 32.f));
  float s, c;
  __sincosf((float)n * fr, &s, &c);
  float x1 = (float)p[j], x2 = (float)p[j + 32];
  p[j]      = (bf16)(x1 * c - x2 * s);
  p[j + 32] = (bf16)(x1 * s + x2 * c);
}

// ---------------------------------------------------------------------------
// Flash attention: one (b,h), 64 queries per block (4 waves x 16 rows),
// 32 keys per iteration, async double-buffered K/V staging, one barrier/iter.
// Softmax scale (0.125) folded into Q at load. Writes ctx bf16 [B,N,D_MODEL].
// ---------------------------------------------------------------------------
__global__ __launch_bounds__(128)
void flash_attn(const float* __restrict__ Q, const bf16* __restrict__ K,
                const bf16* __restrict__ V, bf16* __restrict__ CTX)
{
  // 72-elem rows: 144B (16B-aligned chunks), 36-dword stride -> conflict-free cols
  __shared__ bf16 lds_k[2][32][72];
  __shared__ bf16 lds_v[2][32][72];
  __shared__ bf16 lds_p[4][16][40];    // per-wave P staging: 16 x 32 (+pad)

  const int tid  = threadIdx.x;
  const int wv   = tid >> 5;
  const int lane = tid & 31;
  const int half = lane >> 4;
  const int hl   = lane & 15;
  const int bh   = blockIdx.y;          // b*16 + h
  const int q0   = blockIdx.x * 64;
  const size_t base = (size_t)bh * SEQ * HEAD_DIM;

  auto stage = [&](int kb, int buf) {
    #pragma unroll
    for (int i = 0; i < 2; ++i) {                 // 256 chunks of 8 bf16 each
      int chunk = tid + i * 128;
      int key = chunk >> 3, dc = (chunk & 7) * 8;
      async_copy_b128((unsigned)(uintptr_t)&lds_k[buf][key][dc],
                      K + base + (size_t)(kb + key) * HEAD_DIM + dc);
      async_copy_b128((unsigned)(uintptr_t)&lds_v[buf][key][dc],
                      V + base + (size_t)(kb + key) * HEAD_DIM + dc);
    }
  };

  // preload Q A-fragments for d=[0,32) and d=[32,64), pre-scaled by 1/sqrt(64)
  const int qrow = q0 + wv * 16 + hl;
  const float* qp = Q + base + (size_t)qrow * HEAD_DIM;
  v16bf aq[2];
  #pragma unroll
  for (int c = 0; c < 2; ++c) {
    #pragma unroll
    for (int v = 0; v < 8; ++v) {
      int kk = c * 32 + (v & 3) * 2 + (v >> 2) * 16 + half * 8;
      aq[c][2 * v]     = (bf16)(qp[kk]     * 0.125f);
      aq[c][2 * v + 1] = (bf16)(qp[kk + 1] * 0.125f);
    }
  }

  v8f o[4] = {};
  float mrow[8], lrow[8];
  #pragma unroll
  for (int r = 0; r < 8; ++r) { mrow[r] = -1e30f; lrow[r] = 0.f; }

  stage(0, 0);
  for (int kb = 0; kb < SEQ; kb += 32) {
    const int buf = (kb >> 5) & 1;
    wait_async0();
    __syncthreads();                   // single barrier per iteration
    if (kb + 32 < SEQ) stage(kb + 32, buf ^ 1);

    // ---- S = Q K^T : two 16x16 tiles ----
    v8f s0 = {}, s1 = {};
    #pragma unroll
    for (int c = 0; c < 2; ++c) {
      v16bf b0, b1;
      #pragma unroll
      for (int v = 0; v < 8; ++v) {
        int kk = c * 32 + (v & 3) * 2 + (v >> 2) * 16 + half * 8;     // contraction index (head dim)
        b0[2 * v]     = lds_k[buf][hl][kk];      // B[d][key] = K[key][d]
        b0[2 * v + 1] = lds_k[buf][hl][kk + 1];
        b1[2 * v]     = lds_k[buf][16 + hl][kk];
        b1[2 * v + 1] = lds_k[buf][16 + hl][kk + 1];
      }
      s0 = wmma_bf16(aq[c], b0, s0);
      s1 = wmma_bf16(aq[c], b1, s1);
    }

    // ---- online softmax per row (row = half*8 + r), permlane16 butterflies ----
    #pragma unroll
    for (int r = 0; r < 8; ++r) {
      float v0 = s0[r], v1 = s1[r];
      float mx = reduce_max16(fmaxf(v0, v1));
      float mnew = fmaxf(mrow[r], mx);
      float corr = __expf(mrow[r] - mnew);
      float p0 = __expf(v0 - mnew);
      float p1 = __expf(v1 - mnew);
      float sum = reduce_add16(p0 + p1);
      lrow[r] = lrow[r] * corr + sum;
      mrow[r] = mnew;
      #pragma unroll
      for (int t = 0; t < 4; ++t) o[t][r] *= corr;
      lds_p[wv][half * 8 + r][hl]      = (bf16)p0;   // same-wave LDS RAW in-order
      lds_p[wv][half * 8 + r][16 + hl] = (bf16)p1;
    }

    // ---- O += P V ----
    v16bf ap;
    #pragma unroll
    for (int v = 0; v < 8; ++v) {
      int kk = (v & 3) * 2 + (v >> 2) * 16 + half * 8;
      ap[2 * v]     = lds_p[wv][hl][kk];
      ap[2 * v + 1] = lds_p[wv][hl][kk + 1];
    }
    #pragma unroll
    for (int t = 0; t < 4; ++t) {
      v16bf bv;
      #pragma unroll
      for (int v = 0; v < 8; ++v) {
        int kk = (v & 3) * 2 + (v >> 2) * 16 + half * 8;              // contraction index (key)
        bv[2 * v]     = lds_v[buf][kk][t * 16 + hl];
        bv[2 * v + 1] = lds_v[buf][kk + 1][t * 16 + hl];
      }
      o[t] = wmma_bf16(ap, bv, o[t]);
    }
  }

  // ---- epilogue: normalize, write ctx bf16 [B, N, D_MODEL] ----
  const int b = bh >> 4, h = bh & 15;
  #pragma unroll
  for (int r = 0; r < 8; ++r) {
    float rinv = 1.f / lrow[r];
    int qr = q0 + wv * 16 + half * 8 + r;
    #pragma unroll
    for (int t = 0; t < 4; ++t) {
      int d = t * 16 + hl;
      CTX[((size_t)b * SEQ + qr) * D_MODEL + h * HEAD_DIM + d] =
          (bf16)(o[t][r] * rinv);
    }
  }
}

// ---------------------------------------------------------------------------
// Host side. Workspace layout (~104 MB, assumed <= ws_size):
//   xb   bf16 [8192,1024]  16 MB     Wq/Wk/Wv/Wo bf16 [1024,1024]  2 MB each
//   Q    f32  [B,H,N,64]   32 MB     K,V bf16 [B,H,N,64] 16 MB each
//   CTX  bf16 [B,N,1024]   16 MB
// ---------------------------------------------------------------------------
extern "C" void kernel_launch(void* const* d_in, const int* in_sizes, int n_in,
                              void* d_out, int out_size, void* d_ws, size_t ws_size,
                              hipStream_t stream)
{
  (void)in_sizes; (void)n_in; (void)out_size; (void)ws_size;
  const float* x  = (const float*)d_in[0];
  const float* Wq = (const float*)d_in[1];
  const float* Wk = (const float*)d_in[2];
  const float* Wv = (const float*)d_in[3];
  const float* Wo = (const float*)d_in[4];
  float* out = (float*)d_out;

  char* ws = (char*)d_ws;
  const size_t XE  = (size_t)MROWS * D_MODEL;            // 8M elems
  const size_t WE  = (size_t)D_MODEL * D_MODEL;          // 1M elems
  const size_t QKV = (size_t)BATCH * N_HEADS * SEQ * HEAD_DIM;  // 8M elems

  bf16*  xb  = (bf16*)ws;                     ws += XE * sizeof(bf16);
  bf16*  Wqb = (bf16*)ws;                     ws += WE * sizeof(bf16);
  bf16*  Wkb = (bf16*)ws;                     ws += WE * sizeof(bf16);
  bf16*  Wvb = (bf16*)ws;                     ws += WE * sizeof(bf16);
  bf16*  Wob = (bf16*)ws;                     ws += WE * sizeof(bf16);
  float* Q   = (float*)ws;                    ws += QKV * sizeof(float);
  bf16*  Kb  = (bf16*)ws;                     ws += QKV * sizeof(bf16);
  bf16*  Vb  = (bf16*)ws;                     ws += QKV * sizeof(bf16);
  bf16*  CTX = (bf16*)ws;

  cvt_bf16<<<XE / 1024, 256, 0, stream>>>(x,  xb);
  cvt_bf16<<<WE / 1024, 256, 0, stream>>>(Wq, Wqb);
  cvt_bf16<<<WE / 1024, 256, 0, stream>>>(Wk, Wkb);
  cvt_bf16<<<WE / 1024, 256, 0, stream>>>(Wv, Wvb);
  cvt_bf16<<<WE / 1024, 256, 0, stream>>>(Wo, Wob);

  dim3 blk(128);
  dim3 gemmGrid(D_MODEL / 64, MROWS / 128);   // 16 x 64

  gemm_wmma<0><<<gemmGrid, blk, 0, stream>>>(xb, Wqb, Q, nullptr);
  gemm_wmma<1><<<gemmGrid, blk, 0, stream>>>(xb, Wkb, nullptr, Kb);
  gemm_wmma<1><<<gemmGrid, blk, 0, stream>>>(xb, Wvb, nullptr, Vb);

  const int rope_threads = BATCH * N_HEADS * SEQ * 32;
  rotary_f32 <<<rope_threads / 256, 256, 0, stream>>>(Q);
  rotary_bf16<<<rope_threads / 256, 256, 0, stream>>>(Kb);

  flash_attn<<<dim3(SEQ / 64, BATCH * N_HEADS), blk, 0, stream>>>(Q, Kb, Vb, CTX);

  gemm_wmma<2><<<gemmGrid, blk, 0, stream>>>(CTX, Wob, out, nullptr);
}
